// GPT3ParallelAttention_23081154249387
// MI455X (gfx1250) — matrix operand order repaired and checked by reading, not compiled, not run on hardware
//
#include <hip/hip_runtime.h>

// ---------------- problem sizes (compile-time constants) ----------------
constexpr int S  = 2048;
constexpr int Bn = 2;
constexpr int H  = 2048;
constexpr int NH = 16;
constexpr int D  = 128;
constexpr int M  = S * Bn;        // 4096 rows for projection GEMMs
constexpr int N_QKV = 3 * H;      // 6144
// scores/NORM_FACTOR * COEFF = scores / sqrt(D) when COEFF==1
constexpr float SCALE = 0.08838834764831845f;   // 1/sqrt(128)

typedef __bf16 bf16;
typedef __attribute__((ext_vector_type(16))) bf16  v16bf;
typedef __attribute__((ext_vector_type(8)))  bf16  v8bf;
typedef __attribute__((ext_vector_type(8)))  float v8f;

__device__ __forceinline__ bf16 f2bf(float f) {
  union { float f; unsigned u; } a; a.f = f;
  unsigned r = (a.u + 0x7FFFu + ((a.u >> 16) & 1u)) >> 16;   // RNE
  union { unsigned short s; bf16 b; } o; o.s = (unsigned short)r;
  return o.b;
}

__device__ __forceinline__ v8f wmma_bf16(v16bf a, v16bf b, v8f c) {
  // (neg_a, A, neg_b, B, c_mod, C, reuse_a, reuse_b)
  return __builtin_amdgcn_wmma_f32_16x16x32_bf16(false, a, false, b, (short)0, c,
                                                 false, false);
}

__device__ __forceinline__ v8f vzero8() {
  v8f z;
#pragma unroll
  for (int i = 0; i < 8; ++i) z[i] = 0.0f;
  return z;
}

// ---------------- fp32 -> bf16 conversion ----------------
__global__ void cvt4_f32_bf16(const float* __restrict__ in, bf16* __restrict__ out, int n4) {
  int i = blockIdx.x * blockDim.x + threadIdx.x;
  if (i >= n4) return;
  float4 v = reinterpret_cast<const float4*>(in)[i];
  union { bf16 h[4]; uint2 u; } o;
  o.h[0] = f2bf(v.x); o.h[1] = f2bf(v.y); o.h[2] = f2bf(v.z); o.h[3] = f2bf(v.w);
  reinterpret_cast<uint2*>(out)[i] = o.u;
}

// ---------------- QKV projection GEMM (NT, 32x64 tile per wave) ----------------
// A: [M, H] bf16 row-major (hidden), W: [3H, H] bf16 row-major (qkv_w)
// C[m, n] = sum_k A[m,k] * W[n,k] + bias[n]; scatter to Q/K [bh][s][d], Vt [bh][d][s]
__global__ void gemm_qkv(const bf16* __restrict__ A, const bf16* __restrict__ W,
                         const float* __restrict__ bias,
                         bf16* __restrict__ Qb, bf16* __restrict__ Kb,
                         bf16* __restrict__ Vtb) {
  const int lane = threadIdx.x;
  const int r = lane & 15, g = lane >> 4;
  const int n0 = blockIdx.x * 64;
  const int m0 = blockIdx.y * 32;

  v8f acc[2][4];
#pragma unroll
  for (int am = 0; am < 2; ++am)
#pragma unroll
    for (int nt = 0; nt < 4; ++nt) acc[am][nt] = vzero8();

  for (int k0 = 0; k0 < H; k0 += 32) {
    v16bf a[2];
#pragma unroll
    for (int am = 0; am < 2; ++am) {
      const bf16* ap = A + (size_t)(m0 + am * 16 + r) * H + k0 + g * 16;
      reinterpret_cast<v8bf*>(&a[am])[0] = *reinterpret_cast<const v8bf*>(ap);
      reinterpret_cast<v8bf*>(&a[am])[1] = *reinterpret_cast<const v8bf*>(ap + 8);
    }
#pragma unroll
    for (int nt = 0; nt < 4; ++nt) {
      v16bf bfr = *reinterpret_cast<const v16bf*>(
          W + (size_t)(n0 + nt * 16 + r) * H + k0 + g * 16);
#pragma unroll
      for (int am = 0; am < 2; ++am) acc[am][nt] = wmma_bf16(a[am], bfr, acc[am][nt]);
    }
  }

  // Epilogue: per-column values hoisted; 64-bit (size_t) offset math inside.
#pragma unroll
  for (int nt = 0; nt < 4; ++nt) {
    const int col = n0 + nt * 16 + r;
    const int h = col / 384, j = col % 384;
    const int sel = j >> 7, d = j & 127;
    const float bia = bias[col];
    bf16* dst0;       // base for the (b=0,h) slice of whichever of Q/K/Vt
    int   sStride;    // element stride when s increments
    if (sel == 0)      { dst0 = Qb  + ((size_t)h * S) * D + d;  sStride = D; }
    else if (sel == 1) { dst0 = Kb  + ((size_t)h * S) * D + d;  sStride = D; }
    else               { dst0 = Vtb + ((size_t)h * D + d) * S;  sStride = 1; }
    const size_t bStep = (size_t)NH * S * D;   // b=1 slice offset (same for all)
#pragma unroll
    for (int am = 0; am < 2; ++am)
#pragma unroll
      for (int vj = 0; vj < 8; ++vj) {
        const int m = m0 + am * 16 + vj + 8 * g;   // C layout: row = vj + 8*g
        const int s = m >> 1, b = m & 1;           // hidden [S,B,H] -> m = s*B+b
        dst0[(size_t)b * bStep + (size_t)s * sStride] =
            f2bf(acc[am][nt][vj] + bia);
      }
  }
}

// ---------------- flash attention: one wave per (b,h, 16-query tile) ----------------
__global__ void attn_kernel(const bf16* __restrict__ Qb, const bf16* __restrict__ Kb,
                            const bf16* __restrict__ Vtb, bf16* __restrict__ ctxb) {
  const int bh = blockIdx.x;          // b*NH + h
  const int b = bh >> 4, h = bh & 15;
  // LPT scheduling: heaviest (largest qt) causal tiles launch first.
  const int qt = (int)gridDim.y - 1 - (int)blockIdx.y;
  const int q0 = qt * 16;
  const int lane = threadIdx.x;
  const int r = lane & 15, g = lane >> 4;

  const bf16* Qh = Qb + (size_t)bh * S * D;
  const bf16* Kh = Kb + (size_t)bh * S * D;
  const bf16* Vh = Vtb + (size_t)bh * D * S;

  __shared__ bf16 Pl[16 * 32];        // 1 KB P tile staging

  // Q fragments covering D=128 (4 K-chunks of 32), A-operand layout
  v16bf qf[4];
  {
    const bf16* qrow = Qh + (size_t)(q0 + r) * D;
#pragma unroll
    for (int c = 0; c < 4; ++c) {
      reinterpret_cast<v8bf*>(&qf[c])[0] =
          *reinterpret_cast<const v8bf*>(qrow + c * 32 + g * 16);
      reinterpret_cast<v8bf*>(&qf[c])[1] =
          *reinterpret_cast<const v8bf*>(qrow + c * 32 + 8 + g * 16);
    }
  }

  v8f ctx[8];
#pragma unroll
  for (int dt = 0; dt < 8; ++dt) ctx[dt] = vzero8();
  float mrow[8], lrow[8];
#pragma unroll
  for (int vj = 0; vj < 8; ++vj) { mrow[vj] = -3.0e38f; lrow[vj] = 0.0f; }

  const int kend = q0 + 16;           // causal: keys <= last query of tile
  for (int k0 = 0; k0 < kend; k0 += 32) {
    // ---- scores: S(16q x 32k) = Q(16xD) * K^T ----
    v8f sc[2];
    sc[0] = vzero8(); sc[1] = vzero8();
#pragma unroll
    for (int t = 0; t < 2; ++t)
#pragma unroll
      for (int c = 0; c < 4; ++c) {
        v16bf kf = *reinterpret_cast<const v16bf*>(
            Kh + (size_t)(k0 + t * 16 + r) * D + c * 32 + g * 16);
        sc[t] = wmma_bf16(qf[c], kf, sc[t]);
      }

    // ---- scale + causal mask ----
#pragma unroll
    for (int t = 0; t < 2; ++t)
#pragma unroll
      for (int vj = 0; vj < 8; ++vj) {
        const int key = k0 + t * 16 + r;
        const int row = q0 + vj + 8 * g;
        float sv = sc[t][vj] * SCALE;
        if (key > row) sv = -10000.0f;
        sc[t][vj] = sv;
      }

    // ---- online softmax (row reductions across 16-lane groups) ----
#pragma unroll
    for (int vj = 0; vj < 8; ++vj) {
      float mx = fmaxf(sc[0][vj], sc[1][vj]);
#pragma unroll
      for (int msk = 8; msk >= 1; msk >>= 1)
        mx = fmaxf(mx, __shfl_xor(mx, msk, 32));
      const float mnew = fmaxf(mrow[vj], mx);
      const float corr = __expf(mrow[vj] - mnew);
      const float p0 = __expf(sc[0][vj] - mnew);
      const float p1 = __expf(sc[1][vj] - mnew);
      float ps = p0 + p1;
#pragma unroll
      for (int msk = 8; msk >= 1; msk >>= 1) ps += __shfl_xor(ps, msk, 32);
      lrow[vj] = lrow[vj] * corr + ps;
      mrow[vj] = mnew;
      sc[0][vj] = p0; sc[1][vj] = p1;
#pragma unroll
      for (int dt = 0; dt < 8; ++dt) ctx[dt][vj] *= corr;
    }

    // ---- P -> LDS (row-major 16x32), then reload in A-fragment layout ----
#pragma unroll
    for (int t = 0; t < 2; ++t)
#pragma unroll
      for (int vj = 0; vj < 8; ++vj)
        Pl[(vj + 8 * g) * 32 + t * 16 + r] = f2bf(sc[t][vj]);
    __syncthreads();

    v16bf pf;
    reinterpret_cast<v8bf*>(&pf)[0] =
        *reinterpret_cast<const v8bf*>(&Pl[r * 32 + g * 16]);
    reinterpret_cast<v8bf*>(&pf)[1] =
        *reinterpret_cast<const v8bf*>(&Pl[r * 32 + 8 + g * 16]);

    // ---- ctx += P(16x32) * V(32xD); V pre-transposed so B loads contiguous ----
#pragma unroll
    for (int dt = 0; dt < 8; ++dt) {
      v16bf vf = *reinterpret_cast<const v16bf*>(
          Vh + (size_t)(dt * 16 + r) * S + k0 + g * 16);
      ctx[dt] = wmma_bf16(pf, vf, ctx[dt]);
    }
  }

  // ---- finalize: ctx /= l, store as [s*B+b, h*D+d] bf16 for the dense GEMM ----
  float inv[8];
#pragma unroll
  for (int vj = 0; vj < 8; ++vj) inv[vj] = 1.0f / lrow[vj];
#pragma unroll
  for (int dt = 0; dt < 8; ++dt)
#pragma unroll
    for (int vj = 0; vj < 8; ++vj) {
      const int sg = q0 + vj + 8 * g;
      const int d = dt * 16 + r;
      ctxb[((size_t)sg * Bn + b) * H + h * D + d] = f2bf(ctx[dt][vj] * inv[vj]);
    }
}

// ---------------- dense output GEMM (NT, 32x64 tile per wave, fp32 out) ----------------
__global__ void gemm_dense(const bf16* __restrict__ A, const bf16* __restrict__ W,
                           float* __restrict__ out) {
  const int lane = threadIdx.x;
  const int r = lane & 15, g = lane >> 4;
  const int n0 = blockIdx.x * 64;
  const int m0 = blockIdx.y * 32;

  v8f acc[2][4];
#pragma unroll
  for (int am = 0; am < 2; ++am)
#pragma unroll
    for (int nt = 0; nt < 4; ++nt) acc[am][nt] = vzero8();

  for (int k0 = 0; k0 < H; k0 += 32) {
    v16bf a[2];
#pragma unroll
    for (int am = 0; am < 2; ++am) {
      const bf16* ap = A + (size_t)(m0 + am * 16 + r) * H + k0 + g * 16;
      reinterpret_cast<v8bf*>(&a[am])[0] = *reinterpret_cast<const v8bf*>(ap);
      reinterpret_cast<v8bf*>(&a[am])[1] = *reinterpret_cast<const v8bf*>(ap + 8);
    }
#pragma unroll
    for (int nt = 0; nt < 4; ++nt) {
      v16bf bfr = *reinterpret_cast<const v16bf*>(
          W + (size_t)(n0 + nt * 16 + r) * H + k0 + g * 16);
#pragma unroll
      for (int am = 0; am < 2; ++am) acc[am][nt] = wmma_bf16(a[am], bfr, acc[am][nt]);
    }
  }

#pragma unroll
  for (int nt = 0; nt < 4; ++nt) {
    const int col = n0 + nt * 16 + r;
    float* op = out + col;
#pragma unroll
    for (int am = 0; am < 2; ++am)
#pragma unroll
      for (int vj = 0; vj < 8; ++vj) {
        const int m = m0 + am * 16 + vj + 8 * g;
        op[(size_t)m * H] = acc[am][nt][vj];
      }
  }
}

// ---------------- tuple second output: dense_b passthrough ----------------
__global__ void copy_bias(const float* __restrict__ in, float* __restrict__ out) {
  int i = blockIdx.x * blockDim.x + threadIdx.x;
  if (i < H) out[i] = in[i];
}

extern "C" void kernel_launch(void* const* d_in, const int* in_sizes, int n_in,
                              void* d_out, int out_size, void* d_ws, size_t ws_size,
                              hipStream_t stream) {
  (void)in_sizes; (void)n_in; (void)out_size; (void)ws_size;
  const float* hidden  = (const float*)d_in[0];
  // d_in[1] = attention_mask (causal; computed analytically, unused)
  const float* qkvw    = (const float*)d_in[2];
  const float* qkvb    = (const float*)d_in[3];
  const float* densew  = (const float*)d_in[4];
  const float* denseb  = (const float*)d_in[5];
  float* out = (float*)d_out;

  bf16* w = (bf16*)d_ws;
  bf16* hbf      = w; w += (size_t)M * H;          // 8.4M
  bf16* qkvwbf   = w; w += (size_t)N_QKV * H;      // 12.6M
  bf16* densewbf = w; w += (size_t)H * H;          // 4.2M
  bf16* Qb       = w; w += (size_t)Bn * NH * S * D;
  bf16* Kb       = w; w += (size_t)Bn * NH * S * D;
  bf16* Vtb      = w; w += (size_t)Bn * NH * S * D;
  bf16* ctxb     = w;                              // 8.4M

  {
    int n4 = M * H / 4;
    cvt4_f32_bf16<<<(n4 + 255) / 256, 256, 0, stream>>>(hidden, hbf, n4);
    n4 = N_QKV * H / 4;
    cvt4_f32_bf16<<<(n4 + 255) / 256, 256, 0, stream>>>(qkvw, qkvwbf, n4);
    n4 = H * H / 4;
    cvt4_f32_bf16<<<(n4 + 255) / 256, 256, 0, stream>>>(densew, densewbf, n4);
  }

  gemm_qkv<<<dim3(N_QKV / 64, M / 32), 32, 0, stream>>>(hbf, qkvwbf, qkvb,
                                                        Qb, Kb, Vtb);
  attn_kernel<<<dim3(Bn * NH, S / 16), 32, 0, stream>>>(Qb, Kb, Vtb, ctxb);
  gemm_dense<<<dim3(H / 64, M / 32), 32, 0, stream>>>(ctxb, densewbf, out);
  copy_bias<<<(H + 255) / 256, 256, 0, stream>>>(denseb, out + (size_t)M * H);
}
